// GatherByGateAutoencoder_9998683865099
// MI455X (gfx1250) — matrix operand order repaired and checked by reading, not compiled, not run on hardware
//
#include <hip/hip_runtime.h>
#include <math.h>

// ---------------------------------------------------------------------------
// Problem constants (from reference)
// ---------------------------------------------------------------------------
#define BATCH      8192
#define INPUT_DIM  768
#define HIDDEN     1024
#define EMBED      768
#define QUANT      128
#define SHAPE_N    16
#define SHAPE_DIM  16
#define TOPK       8

typedef __attribute__((ext_vector_type(16))) _Float16 v16h;
typedef __attribute__((ext_vector_type(8)))  float    v8f;

union Frag {
    v16h v;
    uint4 u[2];
};
union Frag8 {
    uint4 u;
    _Float16 h[8];
};

// ---------------------------------------------------------------------------
// f32 -> f16 conversion, 8 elements / thread (all sizes are multiples of 8)
// ---------------------------------------------------------------------------
__global__ void cvt_f32_to_f16_x8(const float* __restrict__ in,
                                  _Float16* __restrict__ out, int n8) {
    int i = blockIdx.x * blockDim.x + threadIdx.x;
    int stride = gridDim.x * blockDim.x;
    for (; i < n8; i += stride) {
        const float4 a = ((const float4*)in)[2 * i + 0];
        const float4 b = ((const float4*)in)[2 * i + 1];
        Frag8 r;
        r.h[0] = (_Float16)a.x; r.h[1] = (_Float16)a.y;
        r.h[2] = (_Float16)a.z; r.h[3] = (_Float16)a.w;
        r.h[4] = (_Float16)b.x; r.h[5] = (_Float16)b.y;
        r.h[6] = (_Float16)b.z; r.h[7] = (_Float16)b.w;
        ((uint4*)out)[i] = r.u;
    }
}

// ---------------------------------------------------------------------------
// Tiled WMMA GEMM:  out = act( A[MxK] @ W[KxN] + bias ),  A/W f16 row-major.
// Block 256 threads (8 waves, wave32). Block tile 128(M) x 128(N), K-step 32.
// Waves arranged 4(M) x 2(N); wave tile 32x64 -> 8 x v_wmma_f32_16x16x32_f16
// per K-step. LDS double-buffered, one barrier per K-step. Staging values are
// plain scalars (SROA-friendly) consumed before the WMMA block; launch_bounds
// min-waves=2 raises the VGPR budget so nothing spills to scratch.
// EDGE=true only for the N=16 gate head; all other GEMMs are branch-free.
// ---------------------------------------------------------------------------
#define LDS_STRIDE 40   // 32 + 8 f16 pad; 80B rows keep 16B alignment

template<bool SILU, bool OUTF32>
__device__ __forceinline__ void store_tile(const v8f acc, void* outp,
                                           const float* bias, int gmBase,
                                           int gn, int N, int half) {
    if (gn >= N) return;
    const float b = bias ? bias[gn] : 0.0f;
#pragma unroll
    for (int i = 0; i < 8; ++i) {
        const int gm = gmBase + half * 8 + i;
        float v = acc[i] + b;
        if (SILU) v = v / (1.0f + expf(-v));
        if (OUTF32) ((float*)outp)[(size_t)gm * N + gn] = v;
        else        ((_Float16*)outp)[(size_t)gm * N + gn] = (_Float16)v;
    }
}

template<bool SILU, bool OUTF32, bool EDGE>
__global__ __launch_bounds__(256, 2) void gemm_wmma_kernel(
        const _Float16* __restrict__ A, const _Float16* __restrict__ W,
        const float* __restrict__ bias, void* __restrict__ outp,
        int M, int N, int K) {
    __shared__ __align__(16) _Float16 As[2][128 * LDS_STRIDE];
    __shared__ __align__(16) _Float16 Bs[2][128 * LDS_STRIDE];

    const int m0 = blockIdx.y * 128;
    const int n0 = blockIdx.x * 128;
    const int t = threadIdx.x;
    const int lane   = t & 31;
    const int wave   = t >> 5;            // 0..7
    const int wm     = (wave & 3) * 32;   // wave M offset in block tile
    const int wn     = (wave >> 2) * 64;  // wave N offset in block tile
    const int lane16 = lane & 15;
    const int half   = lane >> 4;

    // ---- staging address components (loop-invariant scalars) --------------
    const int arow0 = t >> 2;                 // 0..63
    const int arow1 = (t + 256) >> 2;         // 64..127
    const int aq    = (t & 3) * 8;            // 0/8/16/24
    const int bk0   = t >> 4;                 // 0..15
    const int bk1   = (t + 256) >> 4;         // 16..31
    const int bnq   = (t & 15) * 8;           // 0..120

    // global source pointers (advance by 32 f16 per K-step)
    const _Float16* pa0 = A + (size_t)(m0 + arow0) * K + aq;
    const _Float16* pa1 = A + (size_t)(m0 + arow1) * K + aq;
    const _Float16* pw0 = W + (size_t)bk0 * N + n0 + bnq;
    const _Float16* pw1 = W + (size_t)bk1 * N + n0 + bnq;

    uint4 ra0, ra1;
    Frag8 rb0, rb1;
    Frag  a0, a1, b0, b1, b2, b3;

    auto load_tiles = [&](int kk) {
        ra0 = *(const uint4*)(pa0 + kk);
        ra1 = *(const uint4*)(pa1 + kk);
        if (!EDGE) {
            rb0.u = *(const uint4*)(pw0 + (size_t)kk * N);
            rb1.u = *(const uint4*)(pw1 + (size_t)kk * N);
        } else {
#pragma unroll
            for (int j = 0; j < 8; ++j) {
                const bool in0 = (n0 + bnq + j) < N;
                rb0.h[j] = in0 ? pw0[(size_t)kk * N + j] : (_Float16)0.0f;
                rb1.h[j] = in0 ? pw1[(size_t)kk * N + j] : (_Float16)0.0f;
            }
        }
    };
    auto store_tiles = [&](int buf) {
        *(uint4*)&As[buf][arow0 * LDS_STRIDE + aq] = ra0;
        *(uint4*)&As[buf][arow1 * LDS_STRIDE + aq] = ra1;
#pragma unroll
        for (int j = 0; j < 8; ++j) {
            Bs[buf][(bnq + j) * LDS_STRIDE + bk0] = rb0.h[j];
            Bs[buf][(bnq + j) * LDS_STRIDE + bk1] = rb1.h[j];
        }
    };
    // ---- fragments: A and B share contiguous k order -----------------------------
    // A lanes: M = lane16 ; halves hold K 0..15 / 16..31 (contiguous)
    // B lanes: N = lane16 ; halves hold K 0..15 / 16..31 (contiguous)
    auto load_frags = [&](int buf) {
        const _Float16* p0 = &As[buf][(wm + lane16) * LDS_STRIDE + half * 16];
        a0.u[0] = *(const uint4*)p0;
        a0.u[1] = *(const uint4*)(p0 + 8);
        const _Float16* p1 = &As[buf][(wm + 16 + lane16) * LDS_STRIDE + half * 16];
        a1.u[0] = *(const uint4*)p1;
        a1.u[1] = *(const uint4*)(p1 + 8);
        const _Float16* q0 = &Bs[buf][(wn + lane16) * LDS_STRIDE + half * 16];
        b0.u[0] = *(const uint4*)q0;
        b0.u[1] = *(const uint4*)(q0 + 8);
        const _Float16* q1 = &Bs[buf][(wn + 16 + lane16) * LDS_STRIDE + half * 16];
        b1.u[0] = *(const uint4*)q1;
        b1.u[1] = *(const uint4*)(q1 + 8);
        const _Float16* q2 = &Bs[buf][(wn + 32 + lane16) * LDS_STRIDE + half * 16];
        b2.u[0] = *(const uint4*)q2;
        b2.u[1] = *(const uint4*)(q2 + 8);
        const _Float16* q3 = &Bs[buf][(wn + 48 + lane16) * LDS_STRIDE + half * 16];
        b3.u[0] = *(const uint4*)q3;
        b3.u[1] = *(const uint4*)(q3 + 8);
    };

    v8f acc00 = {}, acc01 = {}, acc02 = {}, acc03 = {};
    v8f acc10 = {}, acc11 = {}, acc12 = {}, acc13 = {};

    auto wmma8 = [&]() {
        acc00 = __builtin_amdgcn_wmma_f32_16x16x32_f16(false, a0.v, false, b0.v,
                                                       (short)0, acc00, false, false);
        acc01 = __builtin_amdgcn_wmma_f32_16x16x32_f16(false, a0.v, false, b1.v,
                                                       (short)0, acc01, false, false);
        acc02 = __builtin_amdgcn_wmma_f32_16x16x32_f16(false, a0.v, false, b2.v,
                                                       (short)0, acc02, false, false);
        acc03 = __builtin_amdgcn_wmma_f32_16x16x32_f16(false, a0.v, false, b3.v,
                                                       (short)0, acc03, false, false);
        acc10 = __builtin_amdgcn_wmma_f32_16x16x32_f16(false, a1.v, false, b0.v,
                                                       (short)0, acc10, false, false);
        acc11 = __builtin_amdgcn_wmma_f32_16x16x32_f16(false, a1.v, false, b1.v,
                                                       (short)0, acc11, false, false);
        acc12 = __builtin_amdgcn_wmma_f32_16x16x32_f16(false, a1.v, false, b2.v,
                                                       (short)0, acc12, false, false);
        acc13 = __builtin_amdgcn_wmma_f32_16x16x32_f16(false, a1.v, false, b3.v,
                                                       (short)0, acc13, false, false);
    };

    // ---- pipeline: preload tile 0 -----------------------------------------
    load_tiles(0);
    store_tiles(0);

    int cur = 0;
    int kk = 0;
    for (; kk + 32 < K; kk += 32) {        // all but last K-step
        __syncthreads();                   // buf[cur] ready; buf[cur^1] free
        load_tiles(kk + 32);               // global loads: issue earliest
        load_frags(cur);                   // 12 x ds_load_b128
        store_tiles(cur ^ 1);              // waits loadcnt; frees ra*/rb*
        wmma8();                           // waits dscnt down to store count
        cur ^= 1;
    }
    // ---- last K-step: no staging ------------------------------------------
    __syncthreads();
    load_frags(cur);
    wmma8();

    store_tile<SILU, OUTF32>(acc00, outp, bias, m0 + wm,      n0 + wn +  0 + lane16, N, half);
    store_tile<SILU, OUTF32>(acc01, outp, bias, m0 + wm,      n0 + wn + 16 + lane16, N, half);
    store_tile<SILU, OUTF32>(acc02, outp, bias, m0 + wm,      n0 + wn + 32 + lane16, N, half);
    store_tile<SILU, OUTF32>(acc03, outp, bias, m0 + wm,      n0 + wn + 48 + lane16, N, half);
    store_tile<SILU, OUTF32>(acc10, outp, bias, m0 + wm + 16, n0 + wn +  0 + lane16, N, half);
    store_tile<SILU, OUTF32>(acc11, outp, bias, m0 + wm + 16, n0 + wn + 16 + lane16, N, half);
    store_tile<SILU, OUTF32>(acc12, outp, bias, m0 + wm + 16, n0 + wn + 32 + lane16, N, half);
    store_tile<SILU, OUTF32>(acc13, outp, bias, m0 + wm + 16, n0 + wn + 48 + lane16, N, half);
}

// ---------------------------------------------------------------------------
// Per-row: top-8 of gate (stable ties -> lowest index), per-group softmax
// threshold, write hot (f32), k (f32), vq (f32), hot/k (f16) for codebook GEMM
// ---------------------------------------------------------------------------
__global__ void topk_threshold_kernel(const _Float16* __restrict__ gate,
                                      const _Float16* __restrict__ enc,
                                      float* __restrict__ out_hot,
                                      float* __restrict__ out_k,
                                      float* __restrict__ out_vq,
                                      _Float16* __restrict__ hotn) {
    const int row = blockIdx.x * blockDim.x + threadIdx.x;
    if (row == 0) *out_vq = 0.0f;
    if (row >= BATCH) return;

    float g[SHAPE_N];
#pragma unroll
    for (int i = 0; i < SHAPE_N; ++i)
        g[i] = (float)gate[(size_t)row * SHAPE_N + i];

    unsigned used = 0;
    unsigned long long mask0 = 0, mask1 = 0;
    float cnt = 0.0f;

    for (int j = 0; j < TOPK; ++j) {
        int best = 0;
        float bv = -3.4e38f;
#pragma unroll
        for (int i = 0; i < SHAPE_N; ++i)
            if (!((used >> i) & 1u) && g[i] > bv) { bv = g[i]; best = i; }
        used |= 1u << best;

        float p[SHAPE_DIM];
        float m = -3.4e38f;
#pragma unroll
        for (int q = 0; q < SHAPE_DIM; ++q) {
            p[q] = (float)enc[(size_t)row * (SHAPE_N * SHAPE_DIM) + best * SHAPE_DIM + q];
            m = fmaxf(m, p[q]);
        }
        float s = 0.0f;
#pragma unroll
        for (int q = 0; q < SHAPE_DIM; ++q) { p[q] = expf(p[q] - m); s += p[q]; }
        const float thr = s * (1.0f / (float)SHAPE_DIM);  // prob>1/16 <=> e>s/16
#pragma unroll
        for (int q = 0; q < SHAPE_DIM; ++q) {
            const float h = (p[q] > thr) ? 1.0f : 0.0f;
            out_hot[(size_t)row * QUANT + j * SHAPE_DIM + q] = h;
            cnt += h;
            const int bit = j * SHAPE_DIM + q;
            if (h > 0.5f) {
                if (bit < 64) mask0 |= 1ull << bit;
                else          mask1 |= 1ull << (bit - 64);
            }
        }
    }

    const float k = fminf(fmaxf(cnt, 1.0f), (float)QUANT);
    out_k[row] = k;
    const float inv = 1.0f / k;
#pragma unroll
    for (int i = 0; i < QUANT; ++i) {
        const bool on = (i < 64) ? ((mask0 >> i) & 1ull) : ((mask1 >> (i - 64)) & 1ull);
        hotn[(size_t)row * QUANT + i] = (_Float16)(on ? inv : 0.0f);
    }
}

// ---------------------------------------------------------------------------
// Host-side launcher
// ---------------------------------------------------------------------------
static inline void launch_cvt(const float* in, _Float16* out, int n,
                              hipStream_t stream) {
    const int n8 = n / 8;   // all tensor sizes are multiples of 8
    int blocks = (n8 + 255) / 256;
    if (blocks > 1024) blocks = 1024;
    cvt_f32_to_f16_x8<<<blocks, 256, 0, stream>>>(in, out, n8);
}

extern "C" void kernel_launch(void* const* d_in, const int* in_sizes, int n_in,
                              void* d_out, int out_size, void* d_ws, size_t ws_size,
                              hipStream_t stream) {
    (void)in_sizes; (void)n_in; (void)out_size; (void)ws_size;

    const float* x      = (const float*)d_in[0];
    const float* enc_w1 = (const float*)d_in[1];
    const float* enc_b1 = (const float*)d_in[2];
    const float* enc_w2 = (const float*)d_in[3];
    const float* enc_b2 = (const float*)d_in[4];
    const float* gate_w1 = (const float*)d_in[5];
    const float* gate_b1 = (const float*)d_in[6];
    const float* gate_w2 = (const float*)d_in[7];
    const float* gate_b2 = (const float*)d_in[8];
    const float* cb_w   = (const float*)d_in[9];
    const float* dec_w1 = (const float*)d_in[10];
    const float* dec_b1 = (const float*)d_in[11];
    const float* dec_w2 = (const float*)d_in[12];
    const float* dec_b2 = (const float*)d_in[13];

    // d_out layout: recon | hot | vq | k
    float* out_recon = (float*)d_out;
    float* out_hot   = out_recon + (size_t)BATCH * INPUT_DIM;
    float* out_vq    = out_hot   + (size_t)BATCH * QUANT;
    float* out_k     = out_vq + 1;

    // workspace carve-out (256B aligned), with aliasing:
    //   x16q: x(f16) for enc1, later reused as q(f16) for dec1
    //   act : enc hidden -> gate hidden -> dec hidden (sequential reuse)
    char* ws = (char*)d_ws;
    size_t off = 0;
    auto carve = [&](size_t bytes) -> char* {
        char* p = ws + off;
        off += (bytes + 255) & ~(size_t)255;
        return p;
    };
    _Float16* x16q   = (_Float16*)carve((size_t)BATCH * INPUT_DIM * 2);
    _Float16* act    = (_Float16*)carve((size_t)BATCH * HIDDEN * 2);
    _Float16* enc16  = (_Float16*)carve((size_t)BATCH * SHAPE_N * SHAPE_DIM * 2);
    _Float16* gate16 = (_Float16*)carve((size_t)BATCH * SHAPE_N * 2);
    _Float16* hotn16 = (_Float16*)carve((size_t)BATCH * QUANT * 2);
    _Float16* w_enc1 = (_Float16*)carve((size_t)INPUT_DIM * HIDDEN * 2);
    _Float16* w_enc2 = (_Float16*)carve((size_t)HIDDEN * (SHAPE_N * SHAPE_DIM) * 2);
    _Float16* w_gt1  = (_Float16*)carve((size_t)(SHAPE_N * SHAPE_DIM) * HIDDEN * 2);
    _Float16* w_gt2  = (_Float16*)carve((size_t)HIDDEN * SHAPE_N * 2);
    _Float16* w_cb   = (_Float16*)carve((size_t)QUANT * EMBED * 2);
    _Float16* w_dec1 = (_Float16*)carve((size_t)EMBED * HIDDEN * 2);
    _Float16* w_dec2 = (_Float16*)carve((size_t)HIDDEN * INPUT_DIM * 2);

    // ---- convert inputs/weights to f16 ------------------------------------
    launch_cvt(x,      x16q,   BATCH * INPUT_DIM, stream);
    launch_cvt(enc_w1, w_enc1, INPUT_DIM * HIDDEN, stream);
    launch_cvt(enc_w2, w_enc2, HIDDEN * SHAPE_N * SHAPE_DIM, stream);
    launch_cvt(gate_w1, w_gt1, SHAPE_N * SHAPE_DIM * HIDDEN, stream);
    launch_cvt(gate_w2, w_gt2, HIDDEN * SHAPE_N, stream);
    launch_cvt(cb_w,   w_cb,   QUANT * EMBED, stream);
    launch_cvt(dec_w1, w_dec1, EMBED * HIDDEN, stream);
    launch_cvt(dec_w2, w_dec2, HIDDEN * INPUT_DIM, stream);

    const dim3 blk(256);
    auto grid_for = [](int M, int N) { return dim3((N + 127) / 128, (M + 127) / 128); };

    // enc hidden = silu(x @ enc_w1 + enc_b1)
    gemm_wmma_kernel<true, false, false><<<grid_for(BATCH, HIDDEN), blk, 0, stream>>>(
        x16q, w_enc1, enc_b1, act, BATCH, HIDDEN, INPUT_DIM);
    // enc = hidden @ enc_w2 + enc_b2            (B x 256)
    gemm_wmma_kernel<false, false, false><<<grid_for(BATCH, SHAPE_N * SHAPE_DIM), blk, 0, stream>>>(
        act, w_enc2, enc_b2, enc16, BATCH, SHAPE_N * SHAPE_DIM, HIDDEN);
    // gate hidden = silu(enc @ gate_w1 + gate_b1)
    gemm_wmma_kernel<true, false, false><<<grid_for(BATCH, HIDDEN), blk, 0, stream>>>(
        enc16, w_gt1, gate_b1, act, BATCH, HIDDEN, SHAPE_N * SHAPE_DIM);
    // gate = hidden @ gate_w2 + gate_b2         (B x 16)  -- EDGE path
    gemm_wmma_kernel<false, false, true><<<grid_for(BATCH, SHAPE_N), blk, 0, stream>>>(
        act, w_gt2, gate_b2, gate16, BATCH, SHAPE_N, HIDDEN);

    // top-k gather + softmax threshold + k + hot/k
    topk_threshold_kernel<<<BATCH / 256, 256, 0, stream>>>(
        gate16, enc16, out_hot, out_k, out_vq, hotn16);

    // q = hot_n @ cb_w                          (B x 768), reuse x16q buffer
    gemm_wmma_kernel<false, false, false><<<grid_for(BATCH, EMBED), blk, 0, stream>>>(
        hotn16, w_cb, (const float*)nullptr, x16q, BATCH, EMBED, QUANT);
    // dec hidden = silu(q @ dec_w1 + dec_b1)
    gemm_wmma_kernel<true, false, false><<<grid_for(BATCH, HIDDEN), blk, 0, stream>>>(
        x16q, w_dec1, dec_b1, act, BATCH, HIDDEN, EMBED);
    // recon = hidden @ dec_w2 + dec_b2  -> f32 directly into d_out
    gemm_wmma_kernel<false, true, false><<<grid_for(BATCH, INPUT_DIM), blk, 0, stream>>>(
        act, w_dec2, dec_b2, out_recon, BATCH, INPUT_DIM, HIDDEN);
}
